// attention_738734375251
// MI455X (gfx1250) — matrix operand order repaired and checked by reading, not compiled, not run on hardware
//
#include <hip/hip_runtime.h>

typedef __attribute__((ext_vector_type(16))) __bf16 bf16x16;
typedef __attribute__((ext_vector_type(8)))  float  f32x8;
typedef __attribute__((ext_vector_type(4)))  int    v4i;

union ABu { uint4 q[2]; bf16x16 v; };

static __device__ inline f32x8 wmma_bf16(const ABu& a, const ABu& b, f32x8 c) {
    return __builtin_amdgcn_wmma_f32_16x16x32_bf16(
        false, a.v, false, b.v, (short)0, c, false, false);
}

#define NN 8192            // tokens = D*H*W
#define CHQ 96             // folded q/k dim
#define VC 288             // value channels (3C)

// ---------------- CDNA5 async global->LDS copy (guarded) ---------------------
#if __has_builtin(__builtin_amdgcn_global_load_async_to_lds_b128)
#define HAS_ASYNC 1
#else
#define HAS_ASYNC 0
#endif

static __device__ inline void cp16(const char* g, char* l) {
#if HAS_ASYNC
    __builtin_amdgcn_global_load_async_to_lds_b128(
        (__attribute__((address_space(1))) v4i*)g,
        (__attribute__((address_space(3))) v4i*)l, 0, 0);
#else
    *(uint4*)l = *(const uint4*)g;
#endif
}
static __device__ inline void async_wait6() {
#if HAS_ASYNC
#if __has_builtin(__builtin_amdgcn_s_wait_asynccnt)
    __builtin_amdgcn_s_wait_asynccnt(6);
#else
    asm volatile("s_wait_asynccnt 0x6" ::: "memory");
#endif
#endif
}
static __device__ inline void async_wait0() {
#if HAS_ASYNC
#if __has_builtin(__builtin_amdgcn_s_wait_asynccnt)
    __builtin_amdgcn_s_wait_asynccnt(0);
#else
    asm volatile("s_wait_asynccnt 0x0" ::: "memory");
#endif
#endif
}

// ---------------- weight prep: fold W3 (+b3), convert W4/W5/W6 to bf16 -------
__global__ void wprep(const float* __restrict__ W3, const float* __restrict__ b3,
                      const float* __restrict__ W4, const float* __restrict__ W5,
                      const float* __restrict__ W6,
                      __bf16* __restrict__ W3f, float* __restrict__ b3f,
                      __bf16* __restrict__ W4b, __bf16* __restrict__ W5b,
                      __bf16* __restrict__ W6b) {
    int i = blockIdx.x * 256 + threadIdx.x;
    if (i < 96) b3f[i] = b3[i] + b3[i + 96] + b3[i + 192];
    if (i < 96 * 288) {
        int o = i / 288, c = i % 288;
        W3f[i] = (__bf16)(W3[o * 288 + c] + W3[(o + 96) * 288 + c] + W3[(o + 192) * 288 + c]);
        W6b[i] = (__bf16)W6[i];
    }
    int j = i - 96 * 288;
    if (j >= 0 && j < 288 * 288) W4b[j] = (__bf16)W4[j];
    int k2 = j - 288 * 288;
    if (k2 >= 0 && k2 < 96 * 96) W5b[k2] = (__bf16)W5[k2];
}

// ---------------- cat = [W1*context ; W2*[h0;h1]]  (f32, [b][288][N]) --------
__global__ __launch_bounds__(256) void catmix(
    const float* __restrict__ ctx, const float* __restrict__ h0,
    const float* __restrict__ h1,
    const float* __restrict__ W1, const float* __restrict__ b1,
    const float* __restrict__ W2, const float* __restrict__ b2,
    float* __restrict__ cat) {
    int n  = blockIdx.x * 256 + threadIdx.x;   // gridDim.x = 32
    int oc = blockIdx.y % 288;
    int b  = blockIdx.y / 288;
    float a;
    if (oc < 192) {
        a = b1[oc];
        const float* w = W1 + oc * 128;
        const float* x = ctx + (size_t)b * 128 * NN + n;
        #pragma unroll 4
        for (int ic = 0; ic < 128; ++ic) a += w[ic] * x[(size_t)ic * NN];
    } else {
        int o = oc - 192;
        a = b2[o];
        const float* w  = W2 + o * 192;
        const float* x0 = h0 + (size_t)b * 96 * NN + n;
        const float* x1 = h1 + (size_t)b * 96 * NN + n;
        #pragma unroll 4
        for (int ic = 0; ic < 96; ++ic) a += w[ic] * x0[(size_t)ic * NN];
        #pragma unroll 4
        for (int ic = 0; ic < 96; ++ic) a += w[96 + ic] * x1[(size_t)ic * NN];
    }
    cat[((size_t)b * 288 + oc) * NN + n] = a;
}

// ---------------- generic channel-mix WMMA GEMM (compile-time modes) ---------
// out[b][oc][n] = bias[oc] + sum_ic W[oc][ic] * X[b][ic][n]
// BMODE 0: X = f32 [b][IC][N] (staged via LDS as bf16)
// BMODE 1: X = bf16 [b][N][IC] (token-major, direct fragment loads)
// OUTMODE 0: bf16 [b][OC][N];  1: bf16 [b][N][OC];  2: f32 [b][OC][N] + xadd
template<int OUTMODE, int BMODE>
__global__ __launch_bounds__(256) void gemm_wmma(
    const __bf16* __restrict__ Wb, const float* __restrict__ bias,
    const float* __restrict__ Xf, const __bf16* __restrict__ Xb,
    __bf16* __restrict__ outB, float* __restrict__ outF,
    const float* __restrict__ xadd,
    int OC, int IC, int pairs) {
    int tid = threadIdx.x;
    int wv = tid >> 5, lane = tid & 31;
    int l16 = lane & 15, kh = lane >> 4;
    int bid   = blockIdx.x;
    int b     = bid / (128 * pairs);
    int rem   = bid % (128 * pairs);
    int nb    = (rem / pairs) * 64;
    int pair  = rem % pairs;
    int octile = pair * 2 + (wv & 1);
    int nloc   = (wv >> 1) * 16 + l16;      // 0..63
    int n      = nb + nloc;
    int oc16   = octile * 16;

    __shared__ __align__(16) __bf16 ldsX[64 * 32];  // [n][ic] bf16
    f32x8 acc = {};
    int ksteps = IC >> 5;

    // streaming pointers (advance per k-step; no per-iter 64-bit index math)
    const char*  wr   = (const char*)(Wb + (size_t)(oc16 + l16) * IC) + 32 * kh;
    const float* xsrc = nullptr;
    const char*  xbr  = nullptr;
    __bf16* xdst = nullptr;
    if (BMODE == 0) {
        xsrc = Xf + (size_t)b * IC * NN + (size_t)(tid >> 6) * NN + nb + (tid & 63);
        xdst = ldsX + (tid & 63) * 32 + (tid >> 6);
    } else {
        xbr = (const char*)(Xb + ((size_t)b * NN + n) * IC) + 32 * kh;
    }
    const char* xlds = (const char*)ldsX + nloc * 64 + 32 * kh;

    for (int ks = 0; ks < ksteps; ++ks) {
        if (BMODE == 0) {
            #pragma unroll
            for (int r = 0; r < 8; ++r)
                xdst[4 * r] = (__bf16)xsrc[(size_t)(4 * r) * NN];
            __syncthreads();
        }
        ABu a, bf;
        a.q[0] = *(const uint4*)wr;
        a.q[1] = *(const uint4*)(wr + 16);
        wr += 64;
        if (BMODE == 0) {
            bf.q[0] = *(const uint4*)xlds;
            bf.q[1] = *(const uint4*)(xlds + 16);
        } else {
            bf.q[0] = *(const uint4*)xbr;
            bf.q[1] = *(const uint4*)(xbr + 16);
            xbr += 64;
        }
        acc = wmma_bf16(a, bf, acc);
        if (BMODE == 0) {
            __syncthreads();
            xsrc += (size_t)32 * NN;
        }
    }
    float vals[8];
    #pragma unroll
    for (int j = 0; j < 8; ++j) vals[j] = acc[j] + bias[oc16 + 8 * kh + j];
    if (OUTMODE == 0) {
        #pragma unroll
        for (int j = 0; j < 8; ++j)
            outB[((size_t)b * OC + oc16 + 8 * kh + j) * NN + n] = (__bf16)vals[j];
    } else if (OUTMODE == 1) {
        union { uint4 q; __bf16 h[8]; } pk;
        #pragma unroll
        for (int j = 0; j < 8; ++j) pk.h[j] = (__bf16)vals[j];
        *(uint4*)((char*)outB + (((size_t)b * NN + n) * OC + oc16 + 8 * kh) * 2) = pk.q;
    } else {
        #pragma unroll
        for (int j = 0; j < 8; ++j) {
            size_t o = ((size_t)b * OC + oc16 + 8 * kh + j) * NN + n;
            outF[o] = vals[j] + xadd[o];
        }
    }
}

// ---------------- flash attention (async double-buffered K/V staging) --------
// qT,kT: bf16 [b][N][96]; v: bf16 [b][288][N]; res: bf16 [b][N][288]
// LDS chunk per iter: K 32x96 (6KB) + V 288x32 (18KB) = 1536 x 16B = 6/thread
#define CHUNK_B 24576
__global__ __launch_bounds__(256) void attn_flash(
    const __bf16* __restrict__ qT, const __bf16* __restrict__ kT,
    const __bf16* __restrict__ vv, __bf16* __restrict__ res) {
    int tid = threadIdx.x;
    int wv = tid >> 5, lane = tid & 31;
    int l16 = lane & 15, kh = lane >> 4;
    int b    = blockIdx.x >> 6;             // 64 query blocks per batch
    int qrow = (blockIdx.x & 63) * 128 + wv * 16;

    __shared__ __align__(16) char   ldsKV[2][CHUNK_B];   // [buf][K 6KB | V 18KB]
    __shared__ __align__(16) __bf16 ldsP[8][16 * 32];    // per-wave P transpose

    // ---- per-thread prefetch slots: 6 x 16B per chunk
    const char* gp[6];
    int gst[6], loff[6];
    {
        const char* kb0 = (const char*)(kT + (size_t)b * NN * CHQ);
        const char* vb0 = (const char*)(vv + (size_t)b * VC * NN);
        #pragma unroll
        for (int t = 0; t < 6; ++t) {
            int idx = tid + t * 256;            // 0..1535
            loff[t] = idx * 16;
            if (idx < 384) {                    // K region: contiguous 6KB chunk
                gp[t]  = kb0 + idx * 16;
                gst[t] = 32 * CHQ * 2;          // 6144B per 32-key step
            } else {                            // V region: [ch][32 keys]
                int i2 = idx - 384, ch = i2 >> 2, c4 = i2 & 3;
                gp[t]  = vb0 + (size_t)ch * (NN * 2) + c4 * 16;
                gst[t] = 64;                    // 32 keys * 2B per step
            }
        }
    }
    char* lbase = (char*)ldsKV;
    auto issue = [&](int bufsel) {
        char* dst = lbase + bufsel * CHUNK_B;
        #pragma unroll
        for (int t = 0; t < 6; ++t) {
            cp16(gp[t], dst + loff[t]);
            gp[t] += gst[t];
        }
    };

    // ---- q fragments (resident)
    ABu aq[3];
    {
        const char* qr = (const char*)(qT + ((size_t)b * NN + qrow + l16) * CHQ) + 32 * kh;
        #pragma unroll
        for (int s = 0; s < 3; ++s) {
            aq[s].q[0] = *(const uint4*)(qr + 64 * s);
            aq[s].q[1] = *(const uint4*)(qr + 64 * s + 16);
        }
    }
    // ---- all-ones B fragment (for WMMA row sums of P)
    ABu onesf;
    {
        uint4 o4; o4.x = o4.y = o4.z = o4.w = 0x3f803f80u;  // bf16 1.0 pairs
        onesf.q[0] = o4; onesf.q[1] = o4;
    }
    float mrow[8], lrow[8];
    f32x8 acc[18];
    #pragma unroll
    for (int j = 0; j < 8; ++j) { mrow[j] = -3.0e38f; lrow[j] = 0.f; }
    #pragma unroll
    for (int ct = 0; ct < 18; ++ct) acc[ct] = (f32x8){};

    issue(0);
    for (int it = 0; it < NN / 32; ++it) {
        int cur = it & 1;
        if (it + 1 < NN / 32) { issue(cur ^ 1); async_wait6(); }
        else                  { async_wait0(); }
        __syncthreads();
        const char* ldsK = lbase + cur * CHUNK_B;       // [key][96] bf16
        const char* ldsV = ldsK + 6144;                 // [ch][32]  bf16

        // ---- S = q * k^T for 2 key tiles
        f32x8 s0 = {}, s1 = {};
        {
            const char* kr0 = ldsK + l16 * 192 + 32 * kh;
            const char* kr1 = ldsK + (16 + l16) * 192 + 32 * kh;
            #pragma unroll
            for (int s = 0; s < 3; ++s) {
                ABu bk0, bk1;
                bk0.q[0] = *(const uint4*)(kr0 + 64 * s);
                bk0.q[1] = *(const uint4*)(kr0 + 64 * s + 16);
                bk1.q[0] = *(const uint4*)(kr1 + 64 * s);
                bk1.q[1] = *(const uint4*)(kr1 + 64 * s + 16);
                s0 = wmma_bf16(aq[s], bk0, s0);
                s1 = wmma_bf16(aq[s], bk1, s1);
            }
        }

        // ---- row max: stage-outer so 8 bpermutes are in flight per stage
        float tm[8];
        #pragma unroll
        for (int j = 0; j < 8; ++j) tm[j] = fmaxf(s0[j], s1[j]);
        #pragma unroll
        for (int m = 1; m <= 8; m <<= 1)
            #pragma unroll
            for (int j = 0; j < 8; ++j) tm[j] = fmaxf(tm[j], __shfl_xor(tm[j], m, 32));

        // ---- online softmax state (rows j+8*kh, wave-local)
        float scale[8];
        bool rescale = false;
        #pragma unroll
        for (int j = 0; j < 8; ++j) {
            float mnew = fmaxf(mrow[j], tm[j]);
            scale[j] = __expf(mrow[j] - mnew);
            rescale |= (scale[j] != 1.0f);
            s0[j] = __expf(s0[j] - mnew);
            s1[j] = __expf(s1[j] - mnew);
            mrow[j] = mnew;
        }
        if (__any(rescale)) {
            #pragma unroll
            for (int ct = 0; ct < 18; ++ct)
                #pragma unroll
                for (int j = 0; j < 8; ++j) acc[ct][j] *= scale[j];
        }

        // ---- P: C-layout -> A-layout via per-wave LDS transpose
        __bf16* pb = ldsP[wv];
        #pragma unroll
        for (int j = 0; j < 8; ++j) {
            pb[(j + 8 * kh) * 32 + l16]      = (__bf16)s0[j];
            pb[(j + 8 * kh) * 32 + 16 + l16] = (__bf16)s1[j];
        }
        ABu ap;
        const char* pr = (const char*)pb + l16 * 64 + 32 * kh;
        ap.q[0] = *(const uint4*)pr;
        ap.q[1] = *(const uint4*)(pr + 16);

        // ---- row sums of P via WMMA with ones (replaces 32 bpermutes)
        f32x8 rs = wmma_bf16(ap, onesf, (f32x8){});
        #pragma unroll
        for (int j = 0; j < 8; ++j) lrow[j] = lrow[j] * scale[j] + rs[j];

        // ---- acc += P * V  (18 channel tiles)
        const char* vr0 = ldsV + l16 * 64 + 32 * kh;
        #pragma unroll
        for (int ct = 0; ct < 18; ++ct) {
            ABu bv;
            bv.q[0] = *(const uint4*)(vr0 + ct * 1024);
            bv.q[1] = *(const uint4*)(vr0 + ct * 1024 + 16);
            acc[ct] = wmma_bf16(ap, bv, acc[ct]);
        }
        __syncthreads();
    }

    float inv[8];
    #pragma unroll
    for (int j = 0; j < 8; ++j) inv[j] = 1.0f / lrow[j];
    #pragma unroll
    for (int ct = 0; ct < 18; ++ct)
        #pragma unroll
        for (int j = 0; j < 8; ++j)
            res[((size_t)b * NN + qrow + j + 8 * kh) * VC + ct * 16 + l16] =
                (__bf16)(acc[ct][j] * inv[j]);
}

// ---------------- workspace layout ------------------------------------------
#define OFF_CAT  ((size_t)0)                 // f32 cat [2][288][N]; reused as bf16 res [2][N][288]
#define OFF_KS   ((size_t)18874368)          // bf16 ksumT [2][N][96]
#define OFF_V    (OFF_KS + 3145728)          // bf16 v [2][288][N]
#define OFF_Q    (OFF_V + 9437184)           // bf16 qT [2][N][96]
#define OFF_W3F  (OFF_Q + 3145728)
#define OFF_W4B  (OFF_W3F + 96*288*2)
#define OFF_W5B  (OFF_W4B + 288*288*2)
#define OFF_W6B  (OFF_W5B + 96*96*2)
#define OFF_B3F  (OFF_W6B + 96*288*2)

extern "C" void kernel_launch(void* const* d_in, const int* in_sizes, int n_in,
                              void* d_out, int out_size, void* d_ws, size_t ws_size,
                              hipStream_t stream) {
    const float* context = (const float*)d_in[0];
    const float* h0 = (const float*)d_in[1];
    const float* h1 = (const float*)d_in[2];
    const float* x  = (const float*)d_in[3];
    const float* W1 = (const float*)d_in[4];
    const float* b1 = (const float*)d_in[5];
    const float* W2 = (const float*)d_in[6];
    const float* b2 = (const float*)d_in[7];
    const float* W3 = (const float*)d_in[8];
    const float* b3 = (const float*)d_in[9];
    const float* W4 = (const float*)d_in[10];
    const float* b4 = (const float*)d_in[11];
    const float* W5 = (const float*)d_in[12];
    const float* b5 = (const float*)d_in[13];
    const float* W6 = (const float*)d_in[14];
    const float* b6 = (const float*)d_in[15];

    char* ws = (char*)d_ws;
    float*  cat = (float*)(ws + OFF_CAT);
    __bf16* res = (__bf16*)(ws + OFF_CAT);     // overlays cat (cat dead by then)
    __bf16* ksT = (__bf16*)(ws + OFF_KS);
    __bf16* vB  = (__bf16*)(ws + OFF_V);
    __bf16* qT  = (__bf16*)(ws + OFF_Q);
    __bf16* W3f = (__bf16*)(ws + OFF_W3F);
    __bf16* W4b = (__bf16*)(ws + OFF_W4B);
    __bf16* W5b = (__bf16*)(ws + OFF_W5B);
    __bf16* W6b = (__bf16*)(ws + OFF_W6B);
    float*  b3f = (float*)(ws + OFF_B3F);

    wprep<<<468, 256, 0, stream>>>(W3, b3, W4, W5, W6, W3f, b3f, W4b, W5b, W6b);
    catmix<<<dim3(32, 576), 256, 0, stream>>>(context, h0, h1, W1, b1, W2, b2, cat);
    // ksumT = fold(W3) * cat   -> bf16 [b][n][96]
    gemm_wmma<1, 0><<<2 * 128 * 3, 256, 0, stream>>>(W3f, b3f, cat, nullptr, ksT, nullptr, nullptr,
                                                     96, 288, 3);
    // v = W4 * cat             -> bf16 [b][288][n]
    gemm_wmma<0, 0><<<2 * 128 * 9, 256, 0, stream>>>(W4b, b4, cat, nullptr, vB, nullptr, nullptr,
                                                     288, 288, 9);
    // qT = W5 * x              -> bf16 [b][n][96]
    gemm_wmma<1, 0><<<2 * 128 * 3, 256, 0, stream>>>(W5b, b5, x, nullptr, qT, nullptr, nullptr,
                                                     96, 96, 3);
    // res = softmax(q k^T) v   -> bf16 [b][n][288]
    attn_flash<<<2 * 64, 256, 0, stream>>>(qT, ksT, vB, res);
    // out = W6 * res + b6 + x  -> f32 [b][96][n]
    gemm_wmma<2, 1><<<2 * 128 * 3, 256, 0, stream>>>(W6b, b6, nullptr, res, nullptr, (float*)d_out, x,
                                                     96, 288, 3);
}
